// CTRNN_62234076119592
// MI455X (gfx1250) — matrix operand order repaired and checked by reading, not compiled, not run on hardware
//
#include <hip/hip_runtime.h>

typedef __attribute__((ext_vector_type(8)))  _Float16 v8h;
typedef __attribute__((ext_vector_type(16))) _Float16 v16h;
typedef __attribute__((ext_vector_type(8)))  float    v8f;

// Problem sizes (fixed by the reference)
// T=256, B=128, I=512, H=1024 ; M = T*B = 32768

// ---------------------------------------------------------------------------
// Kernel 1: ws[b,h] = hidden[b,:] . W_hh[h,:] + b_ih[h] + b_hh[h]   (128x1024)
// ---------------------------------------------------------------------------
__global__ __launch_bounds__(256) void hterm_kernel(
    const float* __restrict__ hidden, const float* __restrict__ W_hh,
    const float* __restrict__ b_ih, const float* __restrict__ b_hh,
    float* __restrict__ ws)
{
    const int b = blockIdx.x;            // 0..127
    __shared__ float hsh[1024];
    for (int i = threadIdx.x; i < 1024; i += 256)
        hsh[i] = hidden[b * 1024 + i];
    __syncthreads();

    for (int h = threadIdx.x; h < 1024; h += 256) {
        const float4* wrow = (const float4*)(W_hh + h * 1024);
        float acc = 0.0f;
#pragma unroll 4
        for (int j = 0; j < 256; ++j) {
            float4 w4 = wrow[j];
            acc += w4.x * hsh[4 * j + 0] + w4.y * hsh[4 * j + 1] +
                   w4.z * hsh[4 * j + 2] + w4.w * hsh[4 * j + 3];
        }
        ws[b * 1024 + h] = acc + b_ih[h] + b_hh[h];
    }
}

// ---------------------------------------------------------------------------
// Fragment load: two 16-byte LDS chunks -> v16h
// ---------------------------------------------------------------------------
__device__ __forceinline__ v16h ld_frag(const _Float16* p, int o0, int o1)
{
    v8h lo = *(const v8h*)(p + o0);
    v8h hi = *(const v8h*)(p + o1);
    v16h r;
#pragma unroll
    for (int i = 0; i < 8; ++i) { r[i] = lo[i]; r[i + 8] = hi[i]; }
    return r;
}

// ---------------------------------------------------------------------------
// Kernel 2: out[m,h] = relu( X[m,:K=512] * W_ih[h,:]^T  + ws[m&127, h] )
// 128x128 tile / block, 8 wave32s, f16 WMMA with f32 accumulate, BK=32,
// double-buffered LDS. Also writes last-timestep copy.
// ---------------------------------------------------------------------------
#define LDK 40   // padded row stride in halves (20 dwords: conflict-free frags)

__global__ __launch_bounds__(256) void rnn_gemm_kernel(
    const float* __restrict__ x, const float* __restrict__ w,
    const float* __restrict__ add, float* __restrict__ out,
    float* __restrict__ last)
{
    __shared__ __align__(16) _Float16 As[2][128 * LDK];
    __shared__ __align__(16) _Float16 Bs[2][128 * LDK];

    const int tid  = threadIdx.x;
    const int lane = tid & 31;
    const int wave = tid >> 5;
    const int ln   = lane & 15;
    const int hi   = lane >> 4;           // half-wave select
    const int Moff = (wave >> 1) * 32;    // 4 waves along M
    const int Noff = (wave & 1) * 64;     // 2 waves along N
    const int m0   = blockIdx.y * 128;
    const int n0   = blockIdx.x * 128;

    float4 pa[4], pb[4];

    auto gload = [&](int kt) {
        const int k0 = kt * 32;
#pragma unroll
        for (int j = 0; j < 4; ++j) {
            const int q = tid + 256 * j;         // 0..1023
            const int r = q >> 3, c = q & 7;     // 128 rows x 8 float4
            pa[j] = *(const float4*)(x + (m0 + r) * 512 + k0 + c * 4);
            pb[j] = *(const float4*)(w + (n0 + r) * 512 + k0 + c * 4);
        }
    };

    struct __align__(8) H4 { _Float16 v[4]; };
    auto lstore = [&](int buf) {
#pragma unroll
        for (int j = 0; j < 4; ++j) {
            const int q = tid + 256 * j;
            const int r = q >> 3, c = q & 7;
            H4 a, b;
            a.v[0] = (_Float16)pa[j].x; a.v[1] = (_Float16)pa[j].y;
            a.v[2] = (_Float16)pa[j].z; a.v[3] = (_Float16)pa[j].w;
            b.v[0] = (_Float16)pb[j].x; b.v[1] = (_Float16)pb[j].y;
            b.v[2] = (_Float16)pb[j].z; b.v[3] = (_Float16)pb[j].w;
            *(H4*)&As[buf][r * LDK + c * 4] = a;
            *(H4*)&Bs[buf][r * LDK + c * 4] = b;
        }
    };

    v8f C[2][4] = {};

    gload(0);
    lstore(0);
    __syncthreads();

#pragma unroll 1
    for (int kt = 0; kt < 16; ++kt) {
        const int cur = kt & 1;
        if (kt < 15) gload(kt + 1);

        v16h Af[2], Bf[4];
        // A 16x32 f16 layout: lane-lo K{0-7,16-23}, lane-hi K{8-15,24-31}
#pragma unroll
        for (int mi = 0; mi < 2; ++mi)
            Af[mi] = ld_frag(&As[cur][(Moff + mi * 16 + ln) * LDK],
                             hi * 8, hi * 8 + 16);
        // B 32x16 f16 layout: lane-lo K{0-7,16-23}, lane-hi K{8-15,24-31}
#pragma unroll
        for (int ni = 0; ni < 4; ++ni)
            Bf[ni] = ld_frag(&Bs[cur][(Noff + ni * 16 + ln) * LDK],
                             hi * 8, hi * 8 + 16);

#pragma unroll
        for (int mi = 0; mi < 2; ++mi)
#pragma unroll
            for (int ni = 0; ni < 4; ++ni)
                C[mi][ni] = __builtin_amdgcn_wmma_f32_16x16x32_f16(
                    false, Af[mi], false, Bf[ni], (short)0, C[mi][ni],
                    false, false);

        if (kt < 15) {
            __syncthreads();      // all waves done reading buffer cur^1
            lstore(cur ^ 1);
            __syncthreads();      // new tile visible before next compute
        }
    }

    // Epilogue: + ws[m&127, h], relu, store (and last-timestep copy)
#pragma unroll
    for (int mi = 0; mi < 2; ++mi)
#pragma unroll
        for (int ni = 0; ni < 4; ++ni) {
            const int gh = n0 + Noff + ni * 16 + ln;
            const int mb = m0 + Moff + mi * 16 + hi * 8;
#pragma unroll
            for (int v = 0; v < 8; ++v) {
                const int gm = mb + v;
                float r = C[mi][ni][v] + add[(gm & 127) * 1024 + gh];
                r = fmaxf(r, 0.0f);
                out[(size_t)gm * 1024 + gh] = r;
                if (gm >= 32640)  // last timestep (m in [32640, 32768))
                    last[(gm - 32640) * 1024 + gh] = r;
            }
        }
}

// ---------------------------------------------------------------------------
extern "C" void kernel_launch(void* const* d_in, const int* in_sizes, int n_in,
                              void* d_out, int out_size, void* d_ws,
                              size_t ws_size, hipStream_t stream)
{
    const float* x      = (const float*)d_in[0];  // (256,128,512)
    const float* hidden = (const float*)d_in[1];  // (128,1024)
    const float* W_ih   = (const float*)d_in[2];  // (1024,512)
    const float* W_hh   = (const float*)d_in[3];  // (1024,1024)
    const float* b_ih   = (const float*)d_in[4];  // (1024,)
    const float* b_hh   = (const float*)d_in[5];  // (1024,)

    float* out  = (float*)d_out;                       // (256,128,1024)
    float* last = out + (size_t)256 * 128 * 1024;      // (128,1024)
    float* ws   = (float*)d_ws;                        // 128*1024 floats

    hterm_kernel<<<dim3(128), 256, 0, stream>>>(hidden, W_hh, b_ih, b_hh, ws);
    rnn_gemm_kernel<<<dim3(8, 256), 256, 0, stream>>>(x, W_ih, ws, out, last);
}
